// ANIRepresentation_21955872817669
// MI455X (gfx1250) — matrix operand order repaired and checked by reading, not compiled, not run on hardware
//
#include <hip/hip_runtime.h>
#include <hip/hip_bf16.h>
#include <stdint.h>

typedef __attribute__((ext_vector_type(16))) _Float16 v16h;
typedef __attribute__((ext_vector_type(8)))  _Float16 v8h;
typedef __attribute__((ext_vector_type(8)))  float    v8f;

#define N_SPECIES   7
#define N_PAIRS_SP  28                      // 7*8/2
#define N_RBF       16
#define RAD_FEAT    (N_SPECIES * N_RBF)     // 112
#define ANG_SECDIV  32                      // N_SEC * N_DIV
#define ANG_FEAT    (N_PAIRS_SP * ANG_SECDIV) // 896
#define OUT_W       (RAD_FEAT + ANG_FEAT)   // 1008
#define M_ITEMS     120                     // C(16,2) angular items per atom
#define KPAD        128                     // K padded to 4 WMMA chunks of 32
#define FSTRIDE     136                     // LDS row stride (halves): 272 B, 16B-aligned, bank-skewed

// ---------------------------------------------------------------------------
// Zero only the radial region of the output (angular region is fully
// overwritten by the WMMA kernel; radial is accumulated with atomics).
// ---------------------------------------------------------------------------
__global__ void ani_zero_radial(float* __restrict__ out, int n_atoms) {
    int i = blockIdx.x * blockDim.x + threadIdx.x;
    if (i < n_atoms * RAD_FEAT) {
        int atom = i / RAD_FEAT;
        int r    = i - atom * RAD_FEAT;
        out[(size_t)atom * OUT_W + r] = 0.0f;
    }
}

// ---------------------------------------------------------------------------
// Radial AEV: one thread per neighbor pair. 16 RBF values, scatter-added to
// both endpoint atoms (species-of-the-other-atom sub-block). 2M f32 atomics
// into a 3.7MB L2-resident table.
// ---------------------------------------------------------------------------
__global__ void ani_radial(const float* __restrict__ d_ij,
                           const int*   __restrict__ species,
                           const int*   __restrict__ atom12,
                           float*       __restrict__ out,
                           int P) {
    int p = blockIdx.x * blockDim.x + threadIdx.x;
    if (p >= P) return;

    float d  = d_ij[p];
    float fc = (d <= 0.51f)
                 ? 0.5f * (__cosf(d * (float)(3.14159265358979323846 / 0.51)) + 1.0f)
                 : 0.0f;

    int a0 = atom12[p];
    int a1 = atom12[P + p];
    int sp0 = species[a0];
    int sp1 = species[a1];
    float* o0 = out + (size_t)a0 * OUT_W + sp1 * N_RBF;
    float* o1 = out + (size_t)a1 * OUT_W + sp0 * N_RBF;

    #pragma unroll
    for (int t = 0; t < N_RBF; ++t) {
        float dd = d - (0.08f + 0.026875f * (float)t);   // SHFR[t]
        float v  = 0.25f * fc * __expf(-1600.0f * dd * dd);
        atomicAdd(o0 + t, v);
        atomicAdd(o1 + t, v);
    }
}

// ---------------------------------------------------------------------------
// Angular AEV: one atom per 128-thread block.
// Phase 1: threads 0..119 compute the 32 angular features of one pair-of-pairs
//          item + its species-pair class; store features transposed in LDS
//          (feature-major, K-contiguous) as f16, class as a byte.
// Phase 2: segmented scatter-sum as a GEMM:
//          Out[28x32] = OneHot[28x120] x Feat[120x32], padded to 32x128x32.
//          4 waves x 4 K-chunks of v_wmma_f32_16x16x32_f16; the one-hot A
//          fragment is generated in registers from class-byte compares.
// ---------------------------------------------------------------------------
__global__ __launch_bounds__(128)
void ani_angular(const float* __restrict__ r_ij,
                 const float* __restrict__ d_ij,
                 const int*   __restrict__ species,
                 const int*   __restrict__ atom12,
                 const int*   __restrict__ central,
                 const int*   __restrict__ pair12,
                 const int*   __restrict__ sign12,
                 float*       __restrict__ out,
                 int P, int Mtot) {
    __shared__ __align__(16) _Float16     Ft[ANG_SECDIV][FSTRIDE]; // 8704 B
    __shared__ __align__(8)  unsigned char cls[KPAD];              // 128 B

    // cos/sin of SHFZ[s] = (2s+1)*pi/16
    static constexpr float CZ[8] = { 0.98078528f,  0.83146961f,  0.55557023f,  0.19509032f,
                                    -0.19509032f, -0.55557023f, -0.83146961f, -0.98078528f};
    static constexpr float SZ[8] = { 0.19509032f,  0.55557023f,  0.83146961f,  0.98078528f,
                                     0.98078528f,  0.83146961f,  0.55557023f,  0.19509032f};
    static constexpr float SHFA[4] = {0.08f, 0.1475f, 0.215f, 0.2825f};

    const int tid = threadIdx.x;
    const int g0  = blockIdx.x * M_ITEMS;

    // ---------------- Phase 1: per-item feature compute ----------------
    if (tid < M_ITEMS) {
        const int g  = g0 + tid;
        const int p0 = pair12[g];
        const int p1 = pair12[Mtot + g];
        const int s0 = sign12[g];
        const int s1 = sign12[Mtot + g];

        const float d0 = d_ij[p0];
        const float d1 = d_ij[p1];
        const float x0 = r_ij[3 * p0], y0 = r_ij[3 * p0 + 1], z0 = r_ij[3 * p0 + 2];
        const float x1 = r_ij[3 * p1], y1 = r_ij[3 * p1 + 1], z1 = r_ij[3 * p1 + 2];

        // cos(angle) with sign correction folded in as a scalar
        const float dotv = (x0 * x1 + y0 * y1 + z0 * z1) * (float)(s0 * s1);
        const float cosa = 0.95f * dotv / (d0 * d1);
        const float sina = sqrtf(fmaxf(1.0f - cosa * cosa, 0.0f)); // angle in [0,pi]

        const float fc0 = (d0 <= 0.35f)
            ? 0.5f * (__cosf(d0 * (float)(3.14159265358979323846 / 0.35)) + 1.0f) : 0.0f;
        const float fc1 = (d1 <= 0.35f)
            ? 0.5f * (__cosf(d1 * (float)(3.14159265358979323846 / 0.35)) + 1.0f) : 0.0f;
        const float w    = 2.0f * fc0 * fc1;
        const float davg = 0.5f * (d0 + d1);

        float wf2[4];
        #pragma unroll
        for (int q = 0; q < 4; ++q) {
            float t = davg - SHFA[q];
            wf2[q] = w * __expf(-800.0f * t * t);
        }

        #pragma unroll
        for (int s = 0; s < 8; ++s) {
            // cos(angle - SHFZ[s]) = cosa*CZ + sina*SZ  (no arccos needed)
            float c   = 0.5f * (1.0f + cosa * CZ[s] + sina * SZ[s]);
            float c2  = c * c, c4 = c2 * c2, c8 = c4 * c4, c16 = c8 * c8;
            float f1  = c16 * c16;                       // c^32 (ZETA=32)
            #pragma unroll
            for (int q = 0; q < 4; ++q)
                Ft[s * 4 + q][tid] = (_Float16)(f1 * wf2[q]);
        }

        // species-pair class: "other" atom of each leg per sign12
        const int oa0 = atom12[(s0 == 1 ? P : 0) + p0];
        const int oa1 = atom12[(s1 == 1 ? P : 0) + p1];
        const int spa = species[oa0];
        const int spb = species[oa1];
        const int lo  = spa < spb ? spa : spb;
        const int hi  = spa < spb ? spb : spa;
        cls[tid] = (unsigned char)(lo * (13 - lo) / 2 + hi);  // TRIU index, 0..27
    } else {
        // K padding: items 120..127 contribute nothing
        cls[tid] = 0xFF;
        #pragma unroll
        for (int f = 0; f < ANG_SECDIV; ++f) Ft[f][tid] = (_Float16)0.0f;
    }
    __syncthreads();

    // ---------------- Phase 2: Out = OneHot x Feat via WMMA ----------------
    const int wave  = tid >> 5;
    const int lane  = tid & 31;
    const int mbase = (wave >> 1) << 4;   // class-tile row base (0 or 16)
    const int nbase = (wave & 1) << 4;    // feature-tile col base (0 or 16)
    const int hh    = lane >> 4;          // lane half
    const int l15   = lane & 15;
    const int mrow  = mbase + l15;        // A-matrix row held by this lane

    v8f acc = {0.f, 0.f, 0.f, 0.f, 0.f, 0.f, 0.f, 0.f};

    #pragma unroll
    for (int kc = 0; kc < 4; ++kc) {
        const int kbase = kc * 32;

        // A fragment (one-hot), per ISA 16-bit A 16x32 layout:
        //   elems 0..7  -> K = kbase + 8*hh + t
        //   elems 8..15 -> K = kbase + 16 + 8*hh + t
        const uint64_t q0 = *(const uint64_t*)(&cls[kbase + 8 * hh]);
        const uint64_t q1 = *(const uint64_t*)(&cls[kbase + 16 + 8 * hh]);
        v16h a;
        #pragma unroll
        for (int t = 0; t < 8; ++t) {
            a[t]     = ((int)((q0 >> (8 * t)) & 0xFF) == mrow) ? (_Float16)1.0f : (_Float16)0.0f;
            a[t + 8] = ((int)((q1 >> (8 * t)) & 0xFF) == mrow) ? (_Float16)1.0f : (_Float16)0.0f;
        }

        // B fragment: column n = nbase+l15, K at kbase + 8*hh + t and kbase + 16 + 8*hh + t.
        // Ft rows are 272 B apart and koff*2 is a multiple of 16 -> 16B aligned.
        const _Float16* bp = &Ft[nbase + l15][kbase + (hh << 3)];
        const v8h blo = *(const v8h*)(bp);
        const v8h bhi = *(const v8h*)(bp + 16);
        v16h b;
        #pragma unroll
        for (int t = 0; t < 8; ++t) { b[t] = blo[t]; b[t + 8] = bhi[t]; }

        acc = __builtin_amdgcn_wmma_f32_16x16x32_f16(
            /*neg_a=*/false, a, /*neg_b=*/false, b,
            /*c_mod=*/(short)0, acc, /*reuse_a=*/false, /*reuse_b=*/false);
    }

    // D layout: VGPR r -> row mbase + 8*hh + r, col nbase + l15
    const int   catom = central[g0];
    float* ob = out + (size_t)catom * OUT_W + RAD_FEAT;
    #pragma unroll
    for (int r = 0; r < 8; ++r) {
        const int m = mbase + hh * 8 + r;
        if (m < N_PAIRS_SP)
            ob[m * ANG_SECDIV + nbase + l15] = acc[r];
    }
}

// ---------------------------------------------------------------------------
extern "C" void kernel_launch(void* const* d_in, const int* in_sizes, int n_in,
                              void* d_out, int out_size, void* d_ws, size_t ws_size,
                              hipStream_t stream) {
    const float* r_ij    = (const float*)d_in[0];
    const float* d_ij    = (const float*)d_in[1];
    const int*   species = (const int*)  d_in[2];
    const int*   atom12  = (const int*)  d_in[3];
    const int*   central = (const int*)  d_in[4];
    const int*   pair12  = (const int*)  d_in[5];
    const int*   sign12  = (const int*)  d_in[6];
    float*       out     = (float*)d_out;

    const int P    = in_sizes[1];              // #pairs (65536)
    const int N    = in_sizes[2];              // #atoms (8192)
    const int Mtot = in_sizes[4];              // #angular items (983040)
    const int n_ang_blocks = Mtot / M_ITEMS;   // == N

    ani_zero_radial<<<(N * RAD_FEAT + 255) / 256, 256, 0, stream>>>(out, N);
    ani_radial<<<(P + 255) / 256, 256, 0, stream>>>(d_ij, species, atom12, out, P);
    ani_angular<<<n_ang_blocks, 128, 0, stream>>>(r_ij, d_ij, species, atom12,
                                                  central, pair12, sign12, out,
                                                  P, Mtot);
}
